// CompositionalQ_26328149525276
// MI455X (gfx1250) — matrix operand order repaired and checked by reading, not compiled, not run on hardware
//
#include <hip/hip_runtime.h>
#include <stdint.h>

// Problem constants
#define B_TOTAL 65536
#define OBS 96
#define ACTD 32
#define KMIX 8
#define HID 256
#define INP 128           // OBS + ACT
#define M_TILE 128        // rows per workgroup
#define NTILES 16         // 256 / 16 output column tiles
#define BROW 40           // ushorts per B-slice LDS row (32 data + 8 pad -> 80B, 16B aligned)
#define BUFSZ (256 * BROW)        // ushorts per B buffer (20480 B)
#define BUFBYTES (BUFSZ * 2)
#define SLICE_BYTES 16384         // 256 rows * 32 bf16 (unpadded, in global)
#define X1S 136           // x1 LDS row stride in ushorts (128 + 8 pad -> 272B)
#define X2S 264           // x2 LDS row stride in ushorts (256 + 8 pad -> 528B)

typedef __bf16 bfx16 __attribute__((ext_vector_type(16)));
typedef float  v8f   __attribute__((ext_vector_type(8)));
typedef unsigned int v4u __attribute__((ext_vector_type(4)));
typedef int          v8i __attribute__((ext_vector_type(8)));
typedef int          v4i __attribute__((ext_vector_type(4)));

union FragBF { uint4 u[2]; unsigned short s[16]; bfx16 v; };

__device__ __forceinline__ unsigned short f2bf(float f) {
  unsigned x = __builtin_bit_cast(unsigned, f);
  x += 0x7FFFu + ((x >> 16) & 1u);         // round-to-nearest-even
  return (unsigned short)(x >> 16);
}

#if __has_builtin(__builtin_amdgcn_tensor_load_to_lds)
#define USE_TDM 1
#else
#define USE_TDM 0
#endif

#if USE_TDM
// One-shot TDM: contiguous SLICE_BYTES from global -> LDS, with hardware row
// padding: every 16 DWORDs (64B row) insert 4 DWORDs (16B) -> 80B LDS rows.
__device__ __forceinline__ void tdm_load_slice(const void* gsrc, unsigned lds_off) {
  unsigned long long ga = (unsigned long long)(uintptr_t)gsrc;
  v4u g0;
  g0[0] = 1u;                                     // count=1, user descriptor
  g0[1] = lds_off;                                // lds_addr (bytes)
  g0[2] = (unsigned)(ga & 0xFFFFFFFFu);           // global_addr[31:0]
  g0[3] = (unsigned)((ga >> 32) & 0x01FFFFFFu)    // global_addr[56:32]
        | (2u << 30);                             // type = 2 ("image")
  v8i g1;
  g1[0] = (int)((3u << 16)                        // data_size = 8B
              | (1u << 20)                        // pad_enable
              | (3u << 22)                        // pad_interval: 16 DWORDs
              | (3u << 25));                      // pad_amount:   4 DWORDs
  g1[1] = 0;                                      // abar=0, tensor_dim0[15:0]=0
  g1[2] = (int)((1u << 16) | 0x4000u);            // tensor_dim0=2^30, tensor_dim1=1
  g1[3] = (int)((SLICE_BYTES >> 3) << 16);        // tile_dim0 = 2048 elements
  g1[4] = 1;                                      // tile_dim1 = 1
  g1[5] = 0x40000000;                             // tensor_dim0_stride[31:0]
  g1[6] = 0;
  g1[7] = 0;
  v4i z4 = {0, 0, 0, 0};
#if defined(__clang_major__) && (__clang_major__ >= 23)
  v8i z8 = {0, 0, 0, 0, 0, 0, 0, 0};
  __builtin_amdgcn_tensor_load_to_lds(g0, g1, z4, z4, z8, 0);
#else
  __builtin_amdgcn_tensor_load_to_lds(g0, g1, z4, z4, 0);
#endif
}
#endif

// ---------------- weight pre-transform: fp32 [K][I][H] -> bf16 [kstep][n][k32] ----------------
__global__ void prep_w1(const float* __restrict__ W1, unsigned short* __restrict__ W1t) {
  int idx = blockIdx.x * 256 + threadIdx.x;            // 262144 elements
  int kl = idx & 31, n = (idx >> 5) & 255, ksp = idx >> 13;
  int kg = ksp >> 2, i = (ksp & 3) * 32 + kl;          // flat K index = kg*128 + i
  W1t[idx] = f2bf(W1[((size_t)kg * INP + i) * HID + n]);
}
__global__ void prep_w2(const float* __restrict__ W2, unsigned short* __restrict__ W2t) {
  int idx = blockIdx.x * 256 + threadIdx.x;            // 524288 elements
  int kl = idx & 31, n = (idx >> 5) & 255, ksp = idx >> 13;
  int kg = ksp >> 3, i = (ksp & 7) * 32 + kl;          // flat K index = kg*256 + i
  W2t[idx] = f2bf(W2[((size_t)kg * HID + i) * HID + n]);
}

// ---------------- one compositional GEMM layer on WMMA ----------------
// S k-steps of 32; A[b, ks*32+kl] = cw[b, ks>>GSHIFT] * x[b, (ks&GMASK)*32+kl]
template<int S, int GSHIFT, int GMASK, int XSTRIDE>
__device__ __forceinline__ void run_layer(const unsigned short* __restrict__ Wt,
                                          const unsigned short* x_lds,
                                          const float* cw_lds,
                                          unsigned short* Bbuf,
                                          v8f (&acc)[NTILES],
                                          int lane, int wave, int tid) {
  const int hi8  = (lane >> 4) << 3;          // 0 for lanes 0-15, 8 for 16-31
  const int rowA = wave * 16 + (lane & 15);   // A-fragment row (local)

#if USE_TDM
  const unsigned ldsBase = (unsigned)(uintptr_t)(void*)Bbuf;
  if (wave == 0) {
    tdm_load_slice(Wt, ldsBase);                          // slice 0 -> buf 0
    __builtin_amdgcn_s_wait_tensorcnt((short)0);
  }
  __syncthreads();
#else
  uint4 w0, w1, w2, w3;
  { const uint4* g = (const uint4*)(Wt + (size_t)tid * 32);
    w0 = g[0]; w1 = g[1]; w2 = g[2]; w3 = g[3]; }
  { uint4* b = (uint4*)(Bbuf + tid * BROW);
    b[0] = w0; b[1] = w1; b[2] = w2; b[3] = w3; }
  __syncthreads();
#endif

  for (int ks = 0; ks < S; ++ks) {
#if USE_TDM
    if (ks + 1 < S && wave == 0)               // async DMA next slice into other buffer
      tdm_load_slice(Wt + (size_t)(ks + 1) * (SLICE_BYTES / 2),
                     ldsBase + ((unsigned)(ks + 1) & 1u) * BUFBYTES);
#else
    if (ks + 1 < S) {
      const uint4* g = (const uint4*)(Wt + ((size_t)(ks + 1) * 256 + tid) * 32);
      w0 = g[0]; w1 = g[1]; w2 = g[2]; w3 = g[3];
    }
#endif

    // Build A fragment (16x32 bf16): one contiguous 16-element K-run per lane,
    // scaled by the mixture weight via packed bf16 multiply (k-group constant
    // within a 32-wide K-step).
    const float c = cw_lds[rowA * KMIX + (ks >> GSHIFT)];
    const unsigned short* xr = x_lds + rowA * XSTRIDE + (ks & GMASK) * 32 + hi8 * 2;
    FragBF a;
    a.u[0] = *(const uint4*)(xr);
    a.u[1] = *(const uint4*)(xr + 8);
    a.v = a.v * (__bf16)c;                     // v_pk_mul_bf16 x8

    const unsigned short* bcur = Bbuf + (ks & 1) * BUFSZ;
#pragma unroll
    for (int nt = 0; nt < NTILES; ++nt) {
      // B fragment (32x16 bf16): lane holds K=(laneHi?16:0)+0..15 at fixed N
      const unsigned short* bp = bcur + (nt * 16 + (lane & 15)) * BROW + hi8 * 2;
      FragBF bf;
      bf.u[0] = *(const uint4*)(bp);
      bf.u[1] = *(const uint4*)(bp + 8);
      acc[nt] = __builtin_amdgcn_wmma_f32_16x16x32_bf16(
          false, a.v, false, bf.v, (short)0, acc[nt], false, false);
    }

    if (ks + 1 < S) {                          // publish next slice, 1 barrier/step
#if USE_TDM
      if (wave == 0) __builtin_amdgcn_s_wait_tensorcnt((short)0);
#else
      uint4* b = (uint4*)(Bbuf + ((ks + 1) & 1) * BUFSZ + tid * BROW);
      b[0] = w0; b[1] = w1; b[2] = w2; b[3] = w3;
#endif
      __syncthreads();
    }
  }
}

// ---------------- fused main kernel ----------------
__global__ __launch_bounds__(256, 1)
void compq_main(const float* __restrict__ obs, const float* __restrict__ act,
                const float* __restrict__ comp_w,
                const unsigned short* __restrict__ W1t,
                const unsigned short* __restrict__ W2t,
                const float* __restrict__ b1, const float* __restrict__ b2,
                const float* __restrict__ Wout, const float* __restrict__ bout,
                float* __restrict__ out) {
  extern __shared__ unsigned short smem[];
  unsigned short* x1   = smem;                      // 128*136 ushorts
  unsigned short* x2   = x1 + M_TILE * X1S;         // 128*264 ushorts
  unsigned short* Bbuf = x2 + M_TILE * X2S;         // 2 * BUFSZ ushorts
  float*          cw   = (float*)(Bbuf + 2 * BUFSZ);// 128*8 floats

  const int tid = threadIdx.x, lane = tid & 31, wave = tid >> 5;
  const size_t rowBase = (size_t)blockIdx.x * M_TILE;

  // stage comp_w (1024 floats = 1 float4 per thread)
  { const float4* g = (const float4*)(comp_w + rowBase * KMIX);
    ((float4*)cw)[tid] = g[tid]; }
  // stage x = [obs | act] as bf16 (2 threads per row, 64 cols each)
  { int row = tid >> 1, half = tid & 1;
    const float* orow = obs + (rowBase + row) * OBS;
    const float* arow = act + (rowBase + row) * ACTD;
    unsigned short* xr = x1 + row * X1S + half * 64;
#pragma unroll
    for (int c4 = 0; c4 < 64; c4 += 4) {
      int gc = half * 64 + c4;
      float4 v = (gc < OBS) ? *(const float4*)(orow + gc)
                            : *(const float4*)(arow + (gc - OBS));
      ushort4 p; p.x = f2bf(v.x); p.y = f2bf(v.y); p.z = f2bf(v.z); p.w = f2bf(v.w);
      *(ushort4*)(xr + c4) = p;
    }
  }
  __syncthreads();

  v8f acc[NTILES];
  v8f zero = {0.f, 0.f, 0.f, 0.f, 0.f, 0.f, 0.f, 0.f};
#pragma unroll
  for (int nt = 0; nt < NTILES; ++nt) acc[nt] = zero;

  // Layer 1: [128 x 1024] x [1024 x 256], 32 K-steps
  run_layer<32, 2, 3, X1S>(W1t, x1, cw, Bbuf, acc, lane, wave, tid);

  // finalize: + comp_w@b1, ReLU, pack bf16 into x2, zero acc
  const int hi8 = (lane >> 4) << 3;
#pragma unroll
  for (int nt = 0; nt < NTILES; ++nt) {
    int n = nt * 16 + (lane & 15);
    float bc[KMIX];
#pragma unroll
    for (int k = 0; k < KMIX; ++k) bc[k] = __ldg(b1 + (k << 8) + n);
#pragma unroll
    for (int r = 0; r < 8; ++r) {
      int rowl = wave * 16 + hi8 + r;
      float bias = 0.f;
#pragma unroll
      for (int k = 0; k < KMIX; ++k) bias += cw[rowl * KMIX + k] * bc[k];
      float h = fmaxf(acc[nt][r] + bias, 0.f);
      x2[rowl * X2S + n] = f2bf(h);
      acc[nt][r] = 0.f;
    }
  }
  __syncthreads();   // release Bbuf before layer-2 prologue DMA

  // Layer 2: [128 x 2048] x [2048 x 256], 64 K-steps
  run_layer<64, 3, 7, X2S>(W2t, x2, cw, Bbuf, acc, lane, wave, tid);

  // head: relu(h2 + comp_w@b2) . Wout + bout
  float s[8] = {0.f, 0.f, 0.f, 0.f, 0.f, 0.f, 0.f, 0.f};
#pragma unroll
  for (int nt = 0; nt < NTILES; ++nt) {
    int n = nt * 16 + (lane & 15);
    float w = __ldg(Wout + n);
    float bc[KMIX];
#pragma unroll
    for (int k = 0; k < KMIX; ++k) bc[k] = __ldg(b2 + (k << 8) + n);
#pragma unroll
    for (int r = 0; r < 8; ++r) {
      int rowl = wave * 16 + hi8 + r;
      float bias = 0.f;
#pragma unroll
      for (int k = 0; k < KMIX; ++k) bias += cw[rowl * KMIX + k] * bc[k];
      float h = fmaxf(acc[nt][r] + bias, 0.f);
      s[r] += h * w;
    }
  }
  // reduce across the 16-lane halves (C-matrix column lanes)
#pragma unroll
  for (int m = 1; m < 16; m <<= 1)
#pragma unroll
    for (int r = 0; r < 8; ++r) s[r] += __shfl_xor(s[r], m);
  float b0 = __ldg(bout);
#pragma unroll
  for (int r = 0; r < 8; ++r)
    if ((lane & 15) == r)
      out[rowBase + wave * 16 + hi8 + r] = s[r] + b0;
}

extern "C" void kernel_launch(void* const* d_in, const int* in_sizes, int n_in,
                              void* d_out, int out_size, void* d_ws, size_t ws_size,
                              hipStream_t stream) {
  const float* obs    = (const float*)d_in[0];
  const float* act    = (const float*)d_in[1];
  const float* comp_w = (const float*)d_in[2];
  const float* W1     = (const float*)d_in[3];
  const float* b1     = (const float*)d_in[4];
  const float* W2     = (const float*)d_in[5];
  const float* b2     = (const float*)d_in[6];
  const float* Wout   = (const float*)d_in[7];
  const float* bout   = (const float*)d_in[8];

  unsigned short* W1t = (unsigned short*)d_ws;           // 262144 bf16 = 512 KB
  unsigned short* W2t = W1t + KMIX * INP * HID;          // 524288 bf16 = 1 MB

  prep_w1<<<(KMIX * INP * HID) / 256, 256, 0, stream>>>(W1, W1t);
  prep_w2<<<(KMIX * HID * HID) / 256, 256, 0, stream>>>(W2, W2t);

  size_t lds = (size_t)(M_TILE * X1S + M_TILE * X2S + 2 * BUFSZ) * sizeof(unsigned short)
             + (size_t)M_TILE * KMIX * sizeof(float);    // 147456 bytes
  compq_main<<<B_TOTAL / M_TILE, 256, lds, stream>>>(
      obs, act, comp_w, W1t, W2t, b1, b2, Wout, bout, (float*)d_out);
}
